// SCE_87574383165527
// MI455X (gfx1250) — matrix operand order repaired and checked by reading, not compiled, not run on hardware
//
#include <hip/hip_runtime.h>
#include <hip/hip_bf16.h>
#include <math.h>

// Problem dimensions (fixed by the reference)
#define BATCH 2048
#define VDIM  3129
#define DDIM  768
#define VPAD  3136           // 196 * 16, zero-padded rows for WMMA tiles
#define NT    196            // VPAD / 16 tiles per side

typedef __attribute__((ext_vector_type(16))) __bf16 v16bf;
typedef __attribute__((ext_vector_type(8)))  __bf16 v8bf;
typedef __attribute__((ext_vector_type(8)))  float  v8f;

// ---------------------------------------------------------------------------
// 1) Convert Embedding f32 -> bf16, padded to VPAD rows (pad rows = 0).
// ---------------------------------------------------------------------------
__global__ void pad_convert(const float* __restrict__ E, __bf16* __restrict__ Eb) {
    size_t idx = (size_t)blockIdx.x * blockDim.x + threadIdx.x;
    if (idx >= (size_t)VPAD * DDIM) return;
    int row = (int)(idx / DDIM);
    int col = (int)(idx % DDIM);
    float v = (row < VDIM) ? E[(size_t)row * DDIM + col] : 0.0f;
    Eb[idx] = (__bf16)v;
}

// ---------------------------------------------------------------------------
// 2) sim = Eb @ Eb^T via v_wmma_f32_16x16x32_bf16. One wave per 16x16 tile.
//    A fragment (16x32 bf16): lane l holds row M = l&15;
//      K = (l>=16 ? 8:0) + (e<8 ? e : 16+(e-8))  -> two contiguous 16B loads.
//    B fragment (32x16 bf16): lane l holds col N = l&15 (== row N of E);
//      K = (l>=16 ? 8:0) + (e<8 ? e : 16+(e-8))  -> two contiguous 16B loads.
//    C/D (16x16 f32): VGPR r -> row (r + (l>=16?8:0)), col l&15.
// ---------------------------------------------------------------------------
__global__ void gram_wmma(const __bf16* __restrict__ Eb, float* __restrict__ sim) {
    int wave = threadIdx.x >> 5;
    int lane = threadIdx.x & 31;
    int tile = blockIdx.x * (blockDim.x >> 5) + wave;
    if (tile >= NT * NT) return;

    int tM = (tile / NT) * 16;
    int tN = (tile % NT) * 16;
    int l15 = lane & 15;
    int hi  = lane >> 4;       // 0 for lanes 0-15, 1 for lanes 16-31

    const __bf16* arow = Eb + (size_t)(tM + l15) * DDIM + hi * 8;   // K base 0/8
    const __bf16* brow = Eb + (size_t)(tN + l15) * DDIM + hi * 8;   // K base 0/8

    v8f c = {};
    #pragma unroll 4
    for (int k = 0; k < DDIM; k += 32) {
        v8bf a0 = *(const v8bf*)(arow + k);        // K = kb + 0..7
        v8bf a1 = *(const v8bf*)(arow + k + 16);   // K = kb + 16..23
        v8bf b0 = *(const v8bf*)(brow + k);        // K = kb2 + 0..7
        v8bf b1 = *(const v8bf*)(brow + k + 16);   // K = kb2 + 16..23
        v16bf a = __builtin_shufflevector(a0, a1, 0,1,2,3,4,5,6,7,8,9,10,11,12,13,14,15);
        v16bf b = __builtin_shufflevector(b0, b1, 0,1,2,3,4,5,6,7,8,9,10,11,12,13,14,15);
        c = __builtin_amdgcn_wmma_f32_16x16x32_bf16(false, a, false, b,
                                                    (short)0, c, false, false);
    }

    float* out = sim + (size_t)(tM + hi * 8) * VPAD + tN + l15;
    #pragma unroll
    for (int r = 0; r < 8; ++r)
        out[(size_t)r * VPAD] = c[r];
}

// ---------------------------------------------------------------------------
// 3) n[i] = sqrt( sum_j sim[i,j]^2 ), j < VDIM (padded cols are exactly 0).
// ---------------------------------------------------------------------------
__global__ void rownorm(const float* __restrict__ sim, float* __restrict__ nrm) {
    int i = blockIdx.x;
    const float* row = sim + (size_t)i * VPAD;
    float s = 0.0f;
    for (int j = threadIdx.x; j < VDIM; j += blockDim.x) {
        float v = row[j];
        s += v * v;
    }
    __shared__ float red[256];
    red[threadIdx.x] = s;
    __syncthreads();
    for (int off = 128; off > 0; off >>= 1) {
        if (threadIdx.x < off) red[threadIdx.x] += red[threadIdx.x + off];
        __syncthreads();
    }
    if (threadIdx.x == 0) nrm[i] = sqrtf(red[0]);
}

// ---------------------------------------------------------------------------
// 4) mapping[i] = argmax_j ( sim[i,j] / (n_i*n_j + 1e-8) ), diag -> -2,
//    first-occurrence tie-break like jnp.argmax.
// ---------------------------------------------------------------------------
__global__ void argmax_k(const float* __restrict__ sim, const float* __restrict__ nrm,
                         int* __restrict__ mapping) {
    int i = blockIdx.x;
    const float* row = sim + (size_t)i * VPAD;
    float ni = nrm[i];
    float best = -3.0e38f;
    int   bidx = VDIM;
    for (int j = threadIdx.x; j < VDIM; j += blockDim.x) {
        float v = (j == i) ? -2.0f : row[j] / (ni * nrm[j] + 1e-8f);
        if (v > best || (v == best && j < bidx)) { best = v; bidx = j; }
    }
    __shared__ float rv[256];
    __shared__ int   ri[256];
    rv[threadIdx.x] = best;
    ri[threadIdx.x] = bidx;
    __syncthreads();
    for (int off = 128; off > 0; off >>= 1) {
        if (threadIdx.x < off) {
            float ov = rv[threadIdx.x + off];
            int   oi = ri[threadIdx.x + off];
            if (ov > rv[threadIdx.x] || (ov == rv[threadIdx.x] && oi < ri[threadIdx.x])) {
                rv[threadIdx.x] = ov;
                ri[threadIdx.x] = oi;
            }
        }
        __syncthreads();
    }
    if (threadIdx.x == 0) mapping[i] = ri[0];
}

// ---------------------------------------------------------------------------
// 5) Zero semantic buffer and the (poisoned) output accumulator.
// ---------------------------------------------------------------------------
__global__ void zero_k(float* __restrict__ sem, float* __restrict__ out) {
    size_t idx = (size_t)blockIdx.x * blockDim.x + threadIdx.x;
    if (idx < (size_t)BATCH * VDIM) sem[idx] = 0.0f;
    if (idx == 0) out[0] = 0.0f;
}

// ---------------------------------------------------------------------------
// 6) Scatter: sem[i, mapping[g]] = label[i, g] for nonzero labels.
//    Plain stores: duplicate targets resolve arbitrarily, mirroring the
//    unspecified-order duplicate semantics of jnp .at[].set scatter.
// ---------------------------------------------------------------------------
__global__ void scatter_k(const float* __restrict__ label, const int* __restrict__ mapping,
                          float* __restrict__ sem) {
    size_t idx = (size_t)blockIdx.x * blockDim.x + threadIdx.x;
    if (idx >= (size_t)BATCH * VDIM) return;
    int g = (int)(idx % VDIM);
    float l = label[idx];
    if (l != 0.0f) {
        size_t rowbase = idx - (size_t)g;          // i * VDIM
        sem[rowbase + mapping[g]] = l;
    }
}

// ---------------------------------------------------------------------------
// 7) result = (1/B) * sum_{i,j} [ softplus(pred) - pred * sem ]
//    (mean over B*V, times V  ==  sum / B). Stable softplus.
// ---------------------------------------------------------------------------
__global__ void bce_k(const float* __restrict__ pred, const float* __restrict__ sem,
                      float* __restrict__ out) {
    const size_t N = (size_t)BATCH * VDIM;
    size_t stride = (size_t)gridDim.x * blockDim.x;
    float s = 0.0f;
    for (size_t idx = (size_t)blockIdx.x * blockDim.x + threadIdx.x; idx < N; idx += stride) {
        float x = pred[idx];
        float sp = fmaxf(x, 0.0f) + log1pf(expf(-fabsf(x)));   // logaddexp(0, x)
        s += sp - x * sem[idx];
    }
    __shared__ float red[256];
    red[threadIdx.x] = s;
    __syncthreads();
    for (int off = 128; off > 0; off >>= 1) {
        if (threadIdx.x < off) red[threadIdx.x] += red[threadIdx.x + off];
        __syncthreads();
    }
    if (threadIdx.x == 0)
        atomicAdd(out, red[0] * (1.0f / (float)BATCH));
}

// ---------------------------------------------------------------------------
// Launch
// ---------------------------------------------------------------------------
extern "C" void kernel_launch(void* const* d_in, const int* in_sizes, int n_in,
                              void* d_out, int out_size, void* d_ws, size_t ws_size,
                              hipStream_t stream) {
    const float* pred  = (const float*)d_in[0];   // [B, V]
    const float* label = (const float*)d_in[1];   // [B, V]
    const float* E     = (const float*)d_in[2];   // [V, D]
    float* out = (float*)d_out;                   // scalar f32

    // Workspace layout (bytes), all 256B-aligned:
    //   Eb   : VPAD*DDIM  bf16   =  4,816,896
    //   sim  : VPAD*VPAD  f32    = 39,337,984
    //   nrm  : VPAD       f32
    //   map  : VPAD       i32
    //   sem  : BATCH*VDIM f32    = 25,632,768
    char* ws = (char*)d_ws;
    __bf16* Eb  = (__bf16*)(ws);
    float*  sim = (float*)(ws + 4816896);
    float*  nrm = (float*)(ws + 4816896 + 39337984);
    int*    map = (int*)  (ws + 4816896 + 39337984 + 12544);
    float*  sem = (float*)(ws + 4816896 + 39337984 + 12544 + 12544);

    const int T = 256;

    // 1) pad + convert to bf16
    {
        size_t n = (size_t)VPAD * DDIM;
        pad_convert<<<(int)((n + T - 1) / T), T, 0, stream>>>(E, Eb);
    }
    // 2) Gram matrix via WMMA (8 waves/block, 1 wave per 16x16 tile)
    {
        int tiles = NT * NT;                       // 38416
        int wavesPerBlock = T / 32;                // 8
        gram_wmma<<<(tiles + wavesPerBlock - 1) / wavesPerBlock, T, 0, stream>>>(Eb, sim);
    }
    // 3) row norms of sim
    rownorm<<<VDIM, T, 0, stream>>>(sim, nrm);
    // 4) argmax per row -> mapping
    argmax_k<<<VDIM, T, 0, stream>>>(sim, nrm, map);
    // 5) zero semantic + output
    {
        size_t n = (size_t)BATCH * VDIM;
        zero_k<<<(int)((n + T - 1) / T), T, 0, stream>>>(sem, out);
    }
    // 6) scatter labels through mapping
    {
        size_t n = (size_t)BATCH * VDIM;
        scatter_k<<<(int)((n + T - 1) / T), T, 0, stream>>>(label, map, sem);
    }
    // 7) BCE reduction
    bce_k<<<2048, T, 0, stream>>>(pred, sem, out);
}
